// SlotTracker_41652592836907
// MI455X (gfx1250) — matrix operand order repaired and checked by reading, not compiled, not run on hardware
//
#include <hip/hip_runtime.h>
#include <hip/hip_bf16.h>
#include <math.h>

#define BDIM   32
#define NSLOT  64
#define DDIM   256
#define PENALTY 10000.0f
#define TEMP    0.05f
#define NITERS  20

typedef __bf16 bf16_t;
typedef __attribute__((ext_vector_type(16))) __bf16 v16bf;
typedef __attribute__((ext_vector_type(8)))  __bf16 v8bf;
typedef __attribute__((ext_vector_type(8)))  float  v8f;

#define DSTR 264   // bf16 LDS row stride for 256-wide tiles (16B-aligned, bank-skewed)

__device__ __forceinline__ float clip50(float x) {
    return fminf(fmaxf(x, -50.f), 50.f);
}

// ---------------------------------------------------------------------------
// prep: W1c -> Wct[n][k] bf16  (n,k in 0..255),  W2 -> W2t[n][k] bf16 (n<64)
// ---------------------------------------------------------------------------
__global__ void prep_kernel(const float* __restrict__ W1,
                            const float* __restrict__ W2,
                            bf16_t* __restrict__ Wct,
                            bf16_t* __restrict__ W2t) {
    int idx = blockIdx.x * 256 + threadIdx.x;
    if (idx < 256 * 256) {
        int n = idx >> 8, k = idx & 255;
        Wct[n * 256 + k] = (bf16_t)W1[(size_t)(512 + k) * 256 + n]; // diff block of W1
    } else if (idx < 256 * 256 + 64 * 256) {
        int t = idx - 65536;
        int n = t >> 8, k = t & 255;
        W2t[n * 256 + k] = (bf16_t)W2[(size_t)k * 64 + n];
    }
}

// ---------------------------------------------------------------------------
// uv: U[row][n] = b1[n] + sum_k clip(s_t[row][k]) * W1a[k][n]
//     V[row][n] =         sum_k clip(s_t1[row][k]) * W1b[k][n]
// 16 rows per block so W1a/W1b stream from L2 once per 16 rows (16x traffic cut).
// ---------------------------------------------------------------------------
#define UVROWS 16
__global__ void __launch_bounds__(256)
uv_kernel(const float* __restrict__ st,
          const float* __restrict__ st1,
          const float* __restrict__ W1,
          const float* __restrict__ b1,
          float* __restrict__ U,
          float* __restrict__ V) {
    __shared__ float s0[UVROWS][DDIM], s1[UVROWS][DDIM]; // 32 KB
    const int n = threadIdx.x;
    const int base_row = blockIdx.x * UVROWS;

    for (int idx = n; idx < UVROWS * DDIM; idx += 256) {
        int r = idx >> 8, k = idx & 255;
        s0[r][k] = clip50(st [(size_t)(base_row + r) * DDIM + k]);
        s1[r][k] = clip50(st1[(size_t)(base_row + r) * DDIM + k]);
    }
    __syncthreads();

    float au[UVROWS], av[UVROWS];
    const float bias = b1[n];
    #pragma unroll
    for (int r = 0; r < UVROWS; ++r) { au[r] = bias; av[r] = 0.f; }

    #pragma unroll 2
    for (int k = 0; k < DDIM; ++k) {
        const float wa = W1[(size_t)k * 256 + n];
        const float wb = W1[(size_t)(256 + k) * 256 + n];
        #pragma unroll
        for (int r = 0; r < UVROWS; ++r) {
            au[r] = fmaf(s0[r][k], wa, au[r]);
            av[r] = fmaf(s1[r][k], wb, av[r]);
        }
    }
    #pragma unroll
    for (int r = 0; r < UVROWS; ++r) {
        U[(size_t)(base_row + r) * DDIM + n] = au[r];
        V[(size_t)(base_row + r) * DDIM + n] = av[r];
    }
}

// ---------------------------------------------------------------------------
// pair_gemm: one block per (b, i). 256 threads = 8 wave32.
//   GEMM1: |diff| (64x256) @ Wct^T -> h (64x256), + U[i] + V[j], ReLU, bf16
//   GEMM2: h (64x256) @ W2t^T -> h2 (64x64), + b2, ReLU, f32
//   score: h2 @ W3 + b3, penalties, log_alpha out
// B-fragments are batch-loaded into a register array per K-step so the
// global_load clause overlaps the WMMA chain instead of stalling each op.
// ---------------------------------------------------------------------------
__global__ void __launch_bounds__(256)
pair_gemm_kernel(const float* __restrict__ slots_t,
                 const float* __restrict__ slots_t1,
                 const float* __restrict__ alive_t,
                 const float* __restrict__ alive_t1,
                 const bf16_t* __restrict__ Wct,
                 const bf16_t* __restrict__ W2t,
                 const float* __restrict__ U,
                 const float* __restrict__ V,
                 const float* __restrict__ b2,
                 const float* __restrict__ W3,
                 const float* __restrict__ b3,
                 float* __restrict__ LA) {
    __shared__ __align__(16) char smem[2 * NSLOT * DSTR * 2]; // 67584 B
    bf16_t* sdiff = (bf16_t*)smem;                         // region0: 64 x DSTR bf16
    bf16_t* sh    = (bf16_t*)(smem + NSLOT * DSTR * 2);    // region1: 64 x DSTR bf16
    float*  sh2   = (float*)smem;                          // reuse region0: 64 x 65 f32

    const int tid  = threadIdx.x;
    const int bi   = blockIdx.x;
    const int b    = bi >> 6;
    const int i    = bi & 63;
    const int lane = tid & 31;
    const int wv   = tid >> 5;
    const int lrow  = lane & 15;             // M/N index within tile
    const int khalf = (lane >> 4) * 16;      // A-frag K sub-offset (0 or 16)
    const int kbB   = (lane >> 4) * 16;      // B-frag K sub-offset (0 or 16)
    const int jbase = ((lane >> 4) & 1) * 8; // C/D row group offset

    // ---- stage |clip(s_t[i]) - clip(s_t1[j])| as bf16 in LDS ----
    const float* strow = slots_t  + ((size_t)b * NSLOT + i) * DDIM;
    const float* st1b  = slots_t1 + (size_t)b * NSLOT * DDIM;
    for (int idx = tid; idx < NSLOT * DDIM; idx += 256) {
        int j = idx >> 8, k = idx & 255;
        float d = fabsf(clip50(strow[k]) - clip50(st1b[(size_t)j * DDIM + k]));
        sdiff[j * DSTR + k] = (bf16_t)d;
    }
    __syncthreads();

    // ---- GEMM1: wave wv owns rows [mt*16,+16), cols [nh*128,+128) ----
    const int mt = wv >> 1;
    const int nh = wv & 1;
    v8f acc[8];
    #pragma unroll
    for (int nt = 0; nt < 8; ++nt) { v8f z = {}; acc[nt] = z; }

    #pragma unroll
    for (int kt = 0; kt < 8; ++kt) {
        const int k0 = kt * 32;
        // batch-load all B fragments for this K-step first
        v16bf barr[8];
        #pragma unroll
        for (int nt = 0; nt < 8; ++nt) {
            const int n = (nh * 8 + nt) * 16 + lrow;
            barr[nt] = *(const v16bf*)(Wct + (size_t)n * DDIM + k0 + kbB);
        }
        // A fragment from LDS
        const bf16_t* ap = sdiff + (mt * 16 + lrow) * DSTR + k0 + khalf;
        v8bf alo = *(const v8bf*)ap;
        v8bf ahi = *(const v8bf*)(ap + 8);
        v16bf afrag;
        #pragma unroll
        for (int e = 0; e < 8; ++e) { afrag[e] = alo[e]; afrag[e + 8] = ahi[e]; }
        #pragma unroll
        for (int nt = 0; nt < 8; ++nt) {
            acc[nt] = __builtin_amdgcn_wmma_f32_16x16x32_bf16(
                false, afrag, false, barr[nt], (short)0, acc[nt], false, false);
        }
    }

    // epilogue: + U[i][n] + V[j][n], ReLU, bf16 -> sh
    const float* Urow = U + ((size_t)b * NSLOT + i) * DDIM;
    const float* Vb   = V + (size_t)b * NSLOT * DDIM;
    #pragma unroll
    for (int nt = 0; nt < 8; ++nt) {
        const int n = (nh * 8 + nt) * 16 + lrow;
        const float uval = Urow[n];
        #pragma unroll
        for (int r = 0; r < 8; ++r) {
            const int j = mt * 16 + jbase + r;
            float val = acc[nt][r] + uval + Vb[(size_t)j * DDIM + n];
            val = fmaxf(val, 0.f);
            sh[j * DSTR + n] = (bf16_t)val;
        }
    }
    __syncthreads();

    // ---- GEMM2: h(64x256) @ W2t -> 64x64; wave owns 2 tiles ----
    const int ntb = (wv & 1) * 2;
    v8f acc2[2];
    { v8f z = {}; acc2[0] = z; acc2[1] = z; }
    #pragma unroll
    for (int kt = 0; kt < 8; ++kt) {
        const int k0 = kt * 32;
        v16bf barr2[2];
        #pragma unroll
        for (int t = 0; t < 2; ++t) {
            const int n = (ntb + t) * 16 + lrow;
            barr2[t] = *(const v16bf*)(W2t + (size_t)n * DDIM + k0 + kbB);
        }
        const bf16_t* ap = sh + (mt * 16 + lrow) * DSTR + k0 + khalf;
        v8bf alo = *(const v8bf*)ap;
        v8bf ahi = *(const v8bf*)(ap + 8);
        v16bf afrag;
        #pragma unroll
        for (int e = 0; e < 8; ++e) { afrag[e] = alo[e]; afrag[e + 8] = ahi[e]; }
        #pragma unroll
        for (int t = 0; t < 2; ++t) {
            acc2[t] = __builtin_amdgcn_wmma_f32_16x16x32_bf16(
                false, afrag, false, barr2[t], (short)0, acc2[t], false, false);
        }
    }
    #pragma unroll
    for (int t = 0; t < 2; ++t) {
        const int n = (ntb + t) * 16 + lrow;
        const float bias = b2[n];
        #pragma unroll
        for (int r = 0; r < 8; ++r) {
            const int j = mt * 16 + jbase + r;
            sh2[j * 65 + n] = fmaxf(acc2[t][r] + bias, 0.f);
        }
    }
    __syncthreads();

    // ---- score + penalties + log_alpha ----
    if (tid < NSLOT) {
        const int j = tid;
        float s = b3[0];
        #pragma unroll 4
        for (int n = 0; n < 64; ++n) s += sh2[j * 65 + n] * W3[n];
        const float dead_i = (alive_t [b * NSLOT + i] < 0.5f) ? 1.f : 0.f;
        const float dead_j = (alive_t1[b * NSLOT + j] < 0.5f) ? 1.f : 0.f;
        s -= PENALTY * (dead_i + dead_j);
        const float cost = fminf(fmaxf(-s, -PENALTY), PENALTY);
        LA[((size_t)b * NSLOT + i) * NSLOT + j] = -cost / TEMP;
    }
}

// ---------------------------------------------------------------------------
// sinkhorn: one block (256 threads) per batch. 4 lanes cooperate per row/col
// (partial max / sum-exp reduced with shfl_xor inside aligned quads), 20
// row+col passes, then argmax / confidence / matched mask.
// ---------------------------------------------------------------------------
__global__ void __launch_bounds__(256)
sinkhorn_kernel(const float* __restrict__ LA,
                const float* __restrict__ alive_t,
                const float* __restrict__ alive_t1,
                float* __restrict__ out) {
    __shared__ float la[NSLOT * 65];
    const int b   = blockIdx.x;
    const int tid = threadIdx.x;
    const float* src = LA + (size_t)b * NSLOT * NSLOT;
    for (int idx = tid; idx < NSLOT * NSLOT; idx += 256)
        la[(idx >> 6) * 65 + (idx & 63)] = src[idx];
    __syncthreads();

    const int r4 = tid >> 2;        // row (or col) index 0..63
    const int q  = tid & 3;         // quarter within row
    const int c0 = q * 16;

    for (int it = 0; it < NITERS; ++it) {
        { // normalize rows (axis=2)
            float m = -INFINITY;
            #pragma unroll
            for (int jj = 0; jj < 16; ++jj) m = fmaxf(m, la[r4 * 65 + c0 + jj]);
            m = fmaxf(m, __shfl_xor(m, 1));
            m = fmaxf(m, __shfl_xor(m, 2));
            float s = 0.f;
            #pragma unroll
            for (int jj = 0; jj < 16; ++jj) s += __expf(la[r4 * 65 + c0 + jj] - m);
            s += __shfl_xor(s, 1);
            s += __shfl_xor(s, 2);
            const float lse = m + __logf(s);
            #pragma unroll
            for (int jj = 0; jj < 16; ++jj) la[r4 * 65 + c0 + jj] -= lse;
        }
        __syncthreads();
        { // normalize cols (axis=1)
            float m = -INFINITY;
            #pragma unroll
            for (int ii = 0; ii < 16; ++ii) m = fmaxf(m, la[(c0 + ii) * 65 + r4]);
            m = fmaxf(m, __shfl_xor(m, 1));
            m = fmaxf(m, __shfl_xor(m, 2));
            float s = 0.f;
            #pragma unroll
            for (int ii = 0; ii < 16; ++ii) s += __expf(la[(c0 + ii) * 65 + r4] - m);
            s += __shfl_xor(s, 1);
            s += __shfl_xor(s, 2);
            const float lse = m + __logf(s);
            #pragma unroll
            for (int ii = 0; ii < 16; ++ii) la[(c0 + ii) * 65 + r4] -= lse;
        }
        __syncthreads();
    }

    // argmax over row r4 (first-max tie-break, matching jnp.argmax)
    float best = -INFINITY; int bj = 0;
    #pragma unroll
    for (int jj = 0; jj < 16; ++jj) {
        const float v = la[r4 * 65 + c0 + jj];
        if (v > best) { best = v; bj = c0 + jj; }
    }
    #pragma unroll
    for (int mk = 1; mk <= 2; mk <<= 1) {
        const float ov = __shfl_xor(best, mk);
        const int   oj = __shfl_xor(bj, mk);
        if (ov > best || (ov == best && oj < bj)) { best = ov; bj = oj; }
    }
    if (q == 0) {
        const float conf = __expf(best);
        const float at  = (alive_t [b * NSLOT + r4] > 0.5f) ? 1.f : 0.f;
        const float at1 = (alive_t1[b * NSLOT + bj] > 0.5f) ? 1.f : 0.f;
        const float matched = at * at1 * ((conf > 0.3f) ? 1.f : 0.f);
        ((int*)out)[b * NSLOT + r4] = bj;               // perm (int32 bits)
        out[BDIM * NSLOT + b * NSLOT + r4] = matched;   // matched (f32)
    }
}

// ---------------------------------------------------------------------------
extern "C" void kernel_launch(void* const* d_in, const int* in_sizes, int n_in,
                              void* d_out, int out_size, void* d_ws, size_t ws_size,
                              hipStream_t stream) {
    const float* slots_t  = (const float*)d_in[0];
    const float* slots_t1 = (const float*)d_in[1];
    const float* alive_t  = (const float*)d_in[2];
    const float* alive_t1 = (const float*)d_in[3];
    const float* W1 = (const float*)d_in[4];
    const float* b1 = (const float*)d_in[5];
    const float* W2 = (const float*)d_in[6];
    const float* b2 = (const float*)d_in[7];
    const float* W3 = (const float*)d_in[8];
    const float* b3 = (const float*)d_in[9];

    char* ws = (char*)d_ws;
    bf16_t* Wct = (bf16_t*)ws;                                   // 131072 B
    bf16_t* W2t = (bf16_t*)(ws + 131072);                        //  32768 B
    float*  U   = (float*)(ws + 131072 + 32768);                 // 2 MB
    float*  V   = U + (size_t)BDIM * NSLOT * DDIM;               // 2 MB
    float*  LA  = V + (size_t)BDIM * NSLOT * DDIM;               // 512 KB

    prep_kernel<<<320, 256, 0, stream>>>(W1, W2, Wct, W2t);
    uv_kernel<<<(BDIM * NSLOT) / UVROWS, 256, 0, stream>>>(slots_t, slots_t1, W1, b1, U, V);
    pair_gemm_kernel<<<BDIM * NSLOT, 256, 0, stream>>>(
        slots_t, slots_t1, alive_t, alive_t1, Wct, W2t, U, V, b2, W3, b3, LA);
    sinkhorn_kernel<<<BDIM, 256, 0, stream>>>(LA, alive_t, alive_t1, (float*)d_out);
}
